// TopKWindowAttentionLayer_13778255085909
// MI455X (gfx1250) — matrix operand order repaired and checked by reading, not compiled, not run on hardware
//
#include <hip/hip_runtime.h>
#include <math.h>

typedef unsigned short u16;
typedef __attribute__((ext_vector_type(16))) __bf16 v16bf;
typedef __attribute__((ext_vector_type(8)))  float   v8f;

union BV {
  v16bf v;
  uint4 q[2];
  u16   u[16];
};

__device__ __forceinline__ u16 f2bf(float x) {
  unsigned u = __builtin_bit_cast(unsigned, x);
  unsigned r = (u + 0x7FFFu + ((u >> 16) & 1u)) >> 16;
  return (u16)r;
}

__device__ __forceinline__ v8f wmma_bf16(v16bf a, v16bf b, v8f c) {
  return __builtin_amdgcn_wmma_f32_16x16x32_bf16(false, a, false, b, (short)0, c,
                                                 false, false);
}

// 16-bit A operand (16x32): lane L holds row L%16; elements 0-7 = K (L/16)*8..,
// elements 8-15 = K 16+(L/16)*8..  (two contiguous 16B chunks)
__device__ __forceinline__ v16bf load_a(const u16* row_ptr, int hlf) {
  BV r;
  const u16* p = row_ptr + hlf * 8;
  r.q[0] = *(const uint4*)p;
  r.q[1] = *(const uint4*)(p + 16);
  return r.v;
}
// 16-bit B operand (32x16): lane L holds column L%16; elements 0-7 = K (L/16)*8..,
// elements 8-15 = K 16+(L/16)*8..  (two contiguous 16B chunks)
__device__ __forceinline__ v16bf load_b(const u16* col_ptr, int hlf) {
  BV r;
  const u16* p = col_ptr + hlf * 8;
  r.q[0] = *(const uint4*)p;
  r.q[1] = *(const uint4*)(p + 16);
  return r.v;
}

// ---------------------------------------------------------------- cvt f32->bf16
__global__ void cvt_bf16_kernel(const float* __restrict__ in, u16* __restrict__ out, int n) {
  int i = blockIdx.x * 256 + threadIdx.x;
  if (i < n) out[i] = f2bf(in[i]);
}

// ------------------------------------------------- x (b,256,28,28) -> packed bf16
// xw_bf: window-ordered rows (for QKV GEMM A). message_bf[:, 0:256]: raster order.
__global__ void pack_x_kernel(const float* __restrict__ x, u16* __restrict__ xw_bf,
                              u16* __restrict__ message_bf) {
  int b = blockIdx.z, N = blockIdx.x, ch = threadIdx.x;
  float v = x[((size_t)b * 256 + ch) * 784 + N];
  u16 hv = f2bf(v);
  message_bf[((size_t)b * 784 + N) * 512 + ch] = hv;
  int rr = N / 28, cc = N % 28;
  int mm = rr / 7, w1 = rr % 7, nn = cc / 7, w2 = cc % 7;
  int l = (mm * 4 + nn) * 49 + w1 * 7 + w2;
  xw_bf[((size_t)b * 784 + l) * 256 + ch] = hv;
}

// ------------------------------------------------------------------- WMMA GEMM
// C[m][n] = sum_k A[m][k] * W[n][k]  (+bias, optional ReLU, optional bf16 copy)
// One wave computes a 16x64 C tile; 4 waves/block -> 16x256 per block.
// Software-pipelined: operands for step k+32 are issued before consuming step k,
// so each WMMA waits only on a partial load count.
__global__ void __launch_bounds__(128)
gemm_bf16_kernel(const u16* __restrict__ A, const u16* __restrict__ W,
                 const float* __restrict__ bias, float* __restrict__ C,
                 u16* __restrict__ Cbf, int M, int N, int K, int act,
                 size_t Abatch, size_t Cbatch) {
  int lane = threadIdx.x & 31;
  int wid  = threadIdx.x >> 5;
  int r = lane & 15, hlf = lane >> 4;
  int m0 = blockIdx.x * 16;
  int n0 = (blockIdx.y * 4 + wid) * 64;
  int blk = blockIdx.z;
  const u16* Ab = A + (size_t)blk * Abatch;
  float* Cb = C + (size_t)blk * Cbatch;
  u16* Cbfb = Cbf ? (Cbf + (size_t)blk * Cbatch) : (u16*)0;

  const u16* arow = Ab + (size_t)(m0 + r) * K;
  const u16* w0 = W + (size_t)(n0 + r) * K;
  const u16* w1 = W + (size_t)(n0 + 16 + r) * K;
  const u16* w2 = W + (size_t)(n0 + 32 + r) * K;
  const u16* w3 = W + (size_t)(n0 + 48 + r) * K;

  v8f acc0 = {}, acc1 = {}, acc2 = {}, acc3 = {};

  // prologue: stage k = 0
  v16bf a_c  = load_a(arow, hlf);
  v16bf b0_c = load_b(w0, hlf);
  v16bf b1_c = load_b(w1, hlf);
  v16bf b2_c = load_b(w2, hlf);
  v16bf b3_c = load_b(w3, hlf);

  for (int k0 = 0; k0 < K; k0 += 32) {
    v16bf a_u = a_c, b0_u = b0_c, b1_u = b1_c, b2_u = b2_c, b3_u = b3_c;
    int kn = k0 + 32;
    if (kn < K) {                       // issue next-step loads first
      __builtin_prefetch(arow + kn + 32, 0, 3);
      a_c  = load_a(arow + kn, hlf);
      b0_c = load_b(w0 + kn, hlf);
      b1_c = load_b(w1 + kn, hlf);
      b2_c = load_b(w2 + kn, hlf);
      b3_c = load_b(w3 + kn, hlf);
    }
    acc0 = wmma_bf16(a_u, b0_u, acc0);
    acc1 = wmma_bf16(a_u, b1_u, acc1);
    acc2 = wmma_bf16(a_u, b2_u, acc2);
    acc3 = wmma_bf16(a_u, b3_u, acc3);
  }

  v8f acc[4] = {acc0, acc1, acc2, acc3};
  // C layout: element e of lane L -> row m0 + e + 8*(L/16), col = tile_n0 + L%16
#pragma unroll
  for (int j = 0; j < 4; ++j) {
    int cc = n0 + j * 16 + r;
    float bv = bias ? bias[cc] : 0.0f;
#pragma unroll
    for (int e = 0; e < 8; ++e) {
      float vv = acc[j][e] + bv;
      if (act == 1) vv = fmaxf(vv, 0.0f);
      size_t off = (size_t)(m0 + e + 8 * hlf) * N + cc;
      Cb[off] = vv;
      if (Cbfb) Cbfb[off] = f2bf(vv);
    }
  }
}

// -------------------------------------------------------------- window means
__global__ void window_means_kernel(const float* __restrict__ qf, const float* __restrict__ kf,
                                    const float* __restrict__ vf, float* __restrict__ qm,
                                    float* __restrict__ km, float* __restrict__ vm) {
  int b = blockIdx.y, w = blockIdx.x, ch = threadIdx.x;
  size_t base = ((size_t)b * 784 + w * 49) * 256 + ch;
  float sq = 0.f, sk = 0.f, sv = 0.f;
  for (int t = 0; t < 49; ++t) {
    sq += qf[base + (size_t)t * 256];
    sk += kf[base + (size_t)t * 256];
    sv += vf[base + (size_t)t * 256];
  }
  float inv = 1.0f / 49.0f;
  size_t o = ((size_t)b * 16 + w) * 256 + ch;
  qm[o] = sq * inv; km[o] = sk * inv; vm[o] = sv * inv;
}

// ------------------------------------------------------------------ top-8 of 16
__global__ void topk8_kernel(const float* __restrict__ qm, const float* __restrict__ km,
                             int* __restrict__ idx) {
  int t = threadIdx.x;          // 32 threads = (b, window)
  int b = t >> 4, i = t & 15;
  float sim[16];
  for (int j = 0; j < 16; ++j) {
    float s = 0.f;
    for (int c = 0; c < 256; ++c)
      s += qm[((size_t)b * 16 + i) * 256 + c] * km[((size_t)b * 16 + j) * 256 + c];
    sim[j] = s;
  }
  bool used[16] = {};
  for (int k = 0; k < 8; ++k) {
    int best = 0; float bv = -3.0e38f;
    for (int j = 0; j < 16; ++j)
      if (!used[j] && sim[j] > bv) { bv = sim[j]; best = j; }
    used[best] = true;
    idx[(b * 16 + i) * 8 + k] = best;
  }
}

// --------------------------------------------- keys bf16 [b,S,256], V^T [b,h,32,S]
#define S_KEYS 6528
__global__ void build_keys_kernel(const float* __restrict__ kf, const float* __restrict__ vf,
                                  const float* __restrict__ km, const float* __restrict__ vm,
                                  const int* __restrict__ idx, u16* __restrict__ kbf,
                                  u16* __restrict__ vT) {
  int b = blockIdx.y, s = blockIdx.x, ch = threadIdx.x;
  int w = s / 408, t = s % 408;
  float kv, vv;
  if (t < 392) {
    int src = idx[(b * 16 + w) * 8 + t / 49];
    int tok = t % 49;
    size_t p = ((size_t)b * 784 + src * 49 + tok) * 256 + ch;
    kv = kf[p]; vv = vf[p];
  } else {
    int j = t - 392;
    size_t p = ((size_t)b * 16 + j) * 256 + ch;
    kv = km[p]; vv = vm[p];
  }
  kbf[((size_t)b * S_KEYS + s) * 256 + ch] = f2bf(kv);
  int h = ch >> 5, dd = ch & 31;
  vT[(((size_t)b * 8 + h) * 32 + dd) * S_KEYS + s] = f2bf(vv);
}

// ------------------------------------------- flash attention, 1 wave / 16-q tile
// V-tile loads are issued together with the K-tile loads so the softmax
// (LDS round trip + exp) hides their latency before the P*V WMMAs.
__global__ void __launch_bounds__(32)
flash_attn_kernel(const u16* __restrict__ qbf, const u16* __restrict__ kbf,
                  const u16* __restrict__ vT, u16* __restrict__ msgbf) {
  __shared__ float sS[16 * 33];
  __shared__ float sRow[16];
  int lane = threadIdx.x;
  int r = lane & 15, hlf = lane >> 4;
  int t = blockIdx.x, h = blockIdx.y, b = blockIdx.z;

  const u16* qp = qbf + ((size_t)(b * 784) + t * 16 + r) * 256 + h * 32;
  v16bf qa = load_a(qp, hlf);

  const u16* kbase = kbf + (size_t)b * S_KEYS * 256 + h * 32;
  const u16* vrow0 = vT + ((size_t)(b * 8 + h) * 32 + r) * S_KEYS;
  const u16* vrow1 = vT + ((size_t)(b * 8 + h) * 32 + 16 + r) * S_KEYS;

  v8f o0 = {}, o1 = {};
  float mrow = -1.0e30f, lrow = 0.0f;
  const float scale = 0.1767766952966369f;   // 32^-0.5

  for (int s0 = 0; s0 < S_KEYS; s0 += 32) {
    v16bf kb0 = load_b(kbase + (size_t)(s0 + r) * 256, hlf);
    v16bf kb1 = load_b(kbase + (size_t)(s0 + 16 + r) * 256, hlf);
    v16bf vb0 = load_b(vrow0 + s0, hlf);   // independent of softmax: issue early
    v16bf vb1 = load_b(vrow1 + s0, hlf);
    v8f st0 = {}, st1 = {};
    st0 = wmma_bf16(qa, kb0, st0);
    st1 = wmma_bf16(qa, kb1, st1);

    __syncthreads();                       // prev-iter LDS consumers done
#pragma unroll
    for (int e = 0; e < 8; ++e) {
      int rr = e + 8 * hlf;
      sS[rr * 33 + r]      = st0[e] * scale;
      sS[rr * 33 + 16 + r] = st1[e] * scale;
    }
    __syncthreads();

    float p[16];
    float cmax = -1.0e30f;
#pragma unroll
    for (int i = 0; i < 8; ++i) { p[i] = sS[r * 33 + hlf * 8 + i]; cmax = fmaxf(cmax, p[i]); }
#pragma unroll
    for (int i = 0; i < 8; ++i) { p[8 + i] = sS[r * 33 + 16 + hlf * 8 + i]; cmax = fmaxf(cmax, p[8 + i]); }
    cmax = fmaxf(cmax, __shfl_xor(cmax, 16));
    float mnew = fmaxf(mrow, cmax);
    float corr = __expf(mrow - mnew);
    BV pa;
    float psum = 0.0f;
#pragma unroll
    for (int i = 0; i < 16; ++i) {
      float pe = __expf(p[i] - mnew);
      psum += pe;
      pa.u[i] = f2bf(pe);                  // exact A-operand element order
    }
    psum += __shfl_xor(psum, 16);
    lrow = lrow * corr + psum;
    mrow = mnew;
    if (hlf == 0) sRow[r] = corr;
    __syncthreads();
#pragma unroll
    for (int e = 0; e < 8; ++e) {
      float cf = sRow[e + 8 * hlf];
      o0[e] *= cf; o1[e] *= cf;
    }
    o0 = wmma_bf16(pa.v, vb0, o0);
    o1 = wmma_bf16(pa.v, vb1, o1);
  }

  __syncthreads();
  if (hlf == 0) sRow[r] = 1.0f / lrow;
  __syncthreads();
#pragma unroll
  for (int e = 0; e < 8; ++e) {
    int rr = e + 8 * hlf;
    float inv = sRow[rr];
    size_t rowoff = ((size_t)(b * 784) + t * 16 + rr) * 256 + h * 32;
    msgbf[rowoff + r]      = f2bf(o0[e] * inv);
    msgbf[rowoff + 16 + r] = f2bf(o1[e] * inv);
  }
}

// ---------------------------------------------------- LN1 -> message[:, 256:512]
__global__ void ln1_kernel(const float* __restrict__ msgp, const float* __restrict__ g,
                           const float* __restrict__ beta, u16* __restrict__ message_bf) {
  __shared__ float sbuf[256];
  int b = blockIdx.y, N = blockIdx.x, c = threadIdx.x;
  float v = msgp[((size_t)b * 784 + N) * 256 + c];
  sbuf[c] = v; __syncthreads();
  for (int o = 128; o > 0; o >>= 1) { if (c < o) sbuf[c] += sbuf[c + o]; __syncthreads(); }
  float mu = sbuf[0] * (1.0f / 256.0f);
  __syncthreads();
  float d = v - mu;
  sbuf[c] = d * d; __syncthreads();
  for (int o = 128; o > 0; o >>= 1) { if (c < o) sbuf[c] += sbuf[c + o]; __syncthreads(); }
  float var = sbuf[0] * (1.0f / 256.0f);
  float y = d * rsqrtf(var + 1e-5f) * g[c] + beta[c];
  message_bf[((size_t)b * 784 + N) * 512 + 256 + c] = f2bf(y);
}

// ------------------------------------------------------ depthwise 3x3 + GELU
__global__ void dwconv_gelu_kernel(const float* __restrict__ y1, const float* __restrict__ dww,
                                   const float* __restrict__ dwb, u16* __restrict__ ybf) {
  int b = blockIdx.y, N = blockIdx.x, ch = threadIdx.x;   // 512 threads
  int r = N / 28, cc = N % 28;
  float acc = dwb[ch];
#pragma unroll
  for (int dr = -1; dr <= 1; ++dr)
#pragma unroll
    for (int dc = -1; dc <= 1; ++dc) {
      int rr = r + dr, c2 = cc + dc;
      if (rr >= 0 && rr < 28 && c2 >= 0 && c2 < 28)
        acc += y1[((size_t)b * 784 + rr * 28 + c2) * 512 + ch] *
               dww[ch * 9 + (dr + 1) * 3 + (dc + 1)];
    }
  float gv = 0.5f * acc * (1.0f + erff(acc * 0.7071067811865475f));
  ybf[((size_t)b * 784 + N) * 512 + ch] = f2bf(gv);
}

// ------------------------------------------- LN2 + residual + NCHW output store
__global__ void final_kernel(const float* __restrict__ y2, const float* __restrict__ g,
                             const float* __restrict__ beta, const float* __restrict__ x,
                             float* __restrict__ out) {
  __shared__ float sbuf[256];
  int b = blockIdx.y, N = blockIdx.x, c = threadIdx.x;
  float v = y2[((size_t)b * 784 + N) * 256 + c];
  sbuf[c] = v; __syncthreads();
  for (int o = 128; o > 0; o >>= 1) { if (c < o) sbuf[c] += sbuf[c + o]; __syncthreads(); }
  float mu = sbuf[0] * (1.0f / 256.0f);
  __syncthreads();
  float d = v - mu;
  sbuf[c] = d * d; __syncthreads();
  for (int o = 128; o > 0; o >>= 1) { if (c < o) sbuf[c] += sbuf[c + o]; __syncthreads(); }
  float var = sbuf[0] * (1.0f / 256.0f);
  float y = d * rsqrtf(var + 1e-5f) * g[c] + beta[c];
  size_t xo = ((size_t)b * 256 + c) * 784 + N;
  out[xo] = y + x[xo];
}

// =============================================================================
extern "C" void kernel_launch(void* const* d_in, const int* in_sizes, int n_in,
                              void* d_out, int out_size, void* d_ws, size_t ws_size,
                              hipStream_t stream) {
  (void)in_sizes; (void)n_in; (void)out_size; (void)ws_size;
  const float* x     = (const float*)d_in[0];
  const float* Wq    = (const float*)d_in[1];
  const float* Wk    = (const float*)d_in[2];
  const float* Wv    = (const float*)d_in[3];
  const float* Wm    = (const float*)d_in[4];
  const float* ln1_g = (const float*)d_in[5];
  const float* ln1_b = (const float*)d_in[6];
  const float* fc1_w = (const float*)d_in[7];
  const float* fc1_b = (const float*)d_in[8];
  const float* dw_w  = (const float*)d_in[9];
  const float* dw_b  = (const float*)d_in[10];
  const float* fc2_w = (const float*)d_in[11];
  const float* fc2_b = (const float*)d_in[12];
  const float* ln2_g = (const float*)d_in[13];
  const float* ln2_b = (const float*)d_in[14];
  float* out = (float*)d_out;

  char* base = (char*)d_ws;
  size_t off = 0;
  auto carve = [&](size_t bytes) -> void* {
    void* p = base + off;
    off += (bytes + 255) & ~(size_t)255;
    return p;
  };

  u16* wq_bf      = (u16*)carve((size_t)65536 * 2);
  u16* wk_bf      = (u16*)carve((size_t)65536 * 2);
  u16* wv_bf      = (u16*)carve((size_t)65536 * 2);
  u16* wm_bf      = (u16*)carve((size_t)65536 * 2);
  u16* fc1w_bf    = (u16*)carve((size_t)262144 * 2);
  u16* fc2w_bf    = (u16*)carve((size_t)131072 * 2);
  u16* xw_bf      = (u16*)carve((size_t)2 * 784 * 256 * 2);
  u16* message_bf = (u16*)carve((size_t)2 * 784 * 512 * 2);
  float* qf       = (float*)carve((size_t)2 * 784 * 256 * 4);
  float* kf       = (float*)carve((size_t)2 * 784 * 256 * 4);
  float* vf       = (float*)carve((size_t)2 * 784 * 256 * 4);
  u16* qbf        = (u16*)carve((size_t)2 * 784 * 256 * 2);
  float* qm       = (float*)carve((size_t)2 * 16 * 256 * 4);
  float* km       = (float*)carve((size_t)2 * 16 * 256 * 4);
  float* vm       = (float*)carve((size_t)2 * 16 * 256 * 4);
  int* idx        = (int*)carve((size_t)2 * 16 * 8 * 4);
  u16* kkeys      = (u16*)carve((size_t)2 * S_KEYS * 256 * 2);
  u16* vT         = (u16*)carve((size_t)2 * 8 * 32 * S_KEYS * 2);
  u16* msgbf      = (u16*)carve((size_t)2 * 784 * 256 * 2);
  float* msgp     = (float*)carve((size_t)2 * 784 * 256 * 4);
  float* y1       = (float*)carve((size_t)2 * 784 * 512 * 4);
  u16* ybf        = (u16*)carve((size_t)2 * 784 * 512 * 2);
  float* y2       = (float*)carve((size_t)2 * 784 * 256 * 4);

  // 1. weights -> bf16
  cvt_bf16_kernel<<<(65536 + 255) / 256, 256, 0, stream>>>(Wq, wq_bf, 65536);
  cvt_bf16_kernel<<<(65536 + 255) / 256, 256, 0, stream>>>(Wk, wk_bf, 65536);
  cvt_bf16_kernel<<<(65536 + 255) / 256, 256, 0, stream>>>(Wv, wv_bf, 65536);
  cvt_bf16_kernel<<<(65536 + 255) / 256, 256, 0, stream>>>(Wm, wm_bf, 65536);
  cvt_bf16_kernel<<<(262144 + 255) / 256, 256, 0, stream>>>(fc1_w, fc1w_bf, 262144);
  cvt_bf16_kernel<<<(131072 + 255) / 256, 256, 0, stream>>>(fc2_w, fc2w_bf, 131072);

  // 2. pack x
  pack_x_kernel<<<dim3(784, 1, 2), 256, 0, stream>>>(x, xw_bf, message_bf);

  // 3. QKV projections (window-ordered rows)
  gemm_bf16_kernel<<<dim3(49, 1, 2), 128, 0, stream>>>(
      xw_bf, wq_bf, nullptr, qf, qbf, 784, 256, 256, 0, (size_t)784 * 256, (size_t)784 * 256);
  gemm_bf16_kernel<<<dim3(49, 1, 2), 128, 0, stream>>>(
      xw_bf, wk_bf, nullptr, kf, nullptr, 784, 256, 256, 0, (size_t)784 * 256, (size_t)784 * 256);
  gemm_bf16_kernel<<<dim3(49, 1, 2), 128, 0, stream>>>(
      xw_bf, wv_bf, nullptr, vf, nullptr, 784, 256, 256, 0, (size_t)784 * 256, (size_t)784 * 256);

  // 4-5. window means + top-8
  window_means_kernel<<<dim3(16, 2), 256, 0, stream>>>(qf, kf, vf, qm, km, vm);
  topk8_kernel<<<1, 32, 0, stream>>>(qm, km, idx);

  // 6. build key/value sequences
  build_keys_kernel<<<dim3(S_KEYS, 2), 256, 0, stream>>>(kf, vf, km, vm, idx, kkeys, vT);

  // 7. flash attention (msg stays window-ordered, matching the reference concat)
  flash_attn_kernel<<<dim3(49, 8, 2), 32, 0, stream>>>(qbf, kkeys, vT, msgbf);

  // 8. msg @ Wm^T, then LN1 into message[:, 256:512]
  gemm_bf16_kernel<<<dim3(49, 1, 2), 128, 0, stream>>>(
      msgbf, wm_bf, nullptr, msgp, nullptr, 784, 256, 256, 0, (size_t)784 * 256, (size_t)784 * 256);
  ln1_kernel<<<dim3(784, 2), 256, 0, stream>>>(msgp, ln1_g, ln1_b, message_bf);

  // 9. fc1 + ReLU
  gemm_bf16_kernel<<<dim3(49, 2, 2), 128, 0, stream>>>(
      message_bf, fc1w_bf, fc1_b, y1, nullptr, 784, 512, 512, 1, (size_t)784 * 512, (size_t)784 * 512);

  // 10. depthwise conv + GELU
  dwconv_gelu_kernel<<<dim3(784, 2), 512, 0, stream>>>(y1, dw_w, dw_b, ybf);

  // 11. fc2
  gemm_bf16_kernel<<<dim3(49, 1, 2), 128, 0, stream>>>(
      ybf, fc2w_bf, fc2_b, y2, nullptr, 784, 256, 512, 0, (size_t)784 * 512, (size_t)784 * 256);

  // 12. LN2 + residual + NCHW store
  final_kernel<<<dim3(784, 2), 256, 0, stream>>>(y2, ln2_g, ln2_b, x, out);
}
